// nonlocal_block_30906584662665
// MI455X (gfx1250) — matrix operand order repaired and checked by reading, not compiled, not run on hardware
//
#include <hip/hip_runtime.h>

// ---------------------------------------------------------------------------
// Non-local attention block, MI455X (gfx1250, wave32, WMMA bf16 16x16x32).
//   B=4, C=512, H=W=64 -> N=4096, Cqk=64, Cv=128.
// Pipeline:
//   k_qkv : qT/kT/vT (bf16, channel-major) = W{q,k,v} @ x   [WMMA, x read 1x]
//   k_attn: flash attention per 64-query block, fp32 online softmax, bf16 WMMA
//   k_out : y = x + gamma * (Wo @ outT + bo)                [WMMA]
// Workspace (bf16): qT[B*64*N] kT[B*64*N] vT[B*128*N] ov[B*N*128] = 12 MB
// (fits in 192 MB L2 -> attention streams K/V from L2, not HBM).
// All staging uses 128-bit global loads; LDS transposes write packed bf16
// pairs (b32); fragment loads are batched ahead of the WMMA chains.
// ---------------------------------------------------------------------------

typedef __attribute__((ext_vector_type(16))) __bf16 v16bf;
typedef __attribute__((ext_vector_type(8)))  __bf16 v8bf;
typedef __attribute__((ext_vector_type(8)))  float  v8f;

#define CDIM 512
#define NDIM 4096
#define CQK  64
#define CV   128

// pack two f32 -> packed bf16 pair (low = first arg)
__device__ __forceinline__ unsigned pk2f(float a, float b) {
    union { __bf16 h[2]; unsigned u; } p;
    p.h[0] = (__bf16)a; p.h[1] = (__bf16)b;
    return p.u;
}
// pack two bf16 -> u32
__device__ __forceinline__ unsigned pk2b(__bf16 a, __bf16 b) {
    union { __bf16 h[2]; unsigned u; } p;
    p.h[0] = a; p.h[1] = b;
    return p.u;
}

// A-fragment (16x32 bf16, M x K). lane<16: K={0..7,16..23}; lane>=16: K={8..15,24..31}
__device__ __forceinline__ v16bf frag_a(const __bf16* row, int half, int ko) {
    v8bf lo = *(const v8bf*)(row + ko + half * 8);
    v8bf hi = *(const v8bf*)(row + ko + 16 + half * 8);
    v16bf r;
#pragma unroll
    for (int i = 0; i < 8; ++i) { r[i] = lo[i]; r[8 + i] = hi[i]; }
    return r;
}
// B-fragment (32x16 bf16, K x N) from [N][K]-layout LDS row.
// lane<16: K={0..7,16..23}; lane>=16: K={8..15,24..31}
__device__ __forceinline__ v16bf frag_b(const __bf16* row, int half, int ko) {
    v8bf lo = *(const v8bf*)(row + ko + half * 8);
    v8bf hi = *(const v8bf*)(row + ko + 16 + half * 8);
    v16bf r;
#pragma unroll
    for (int i = 0; i < 8; ++i) { r[i] = lo[i]; r[8 + i] = hi[i]; }
    return r;
}
__device__ __forceinline__ v8f wmma_bf16(v16bf a, v16bf b, v8f c) {
    return __builtin_amdgcn_wmma_f32_16x16x32_bf16(false, a, false, b,
                                                   (short)0, c, false, false);
}

// ---------------------------------------------------------------------------
// Kernel 1: fused QKV projection. One block = ALL 256 output channels x 64
// pixels -> x is read exactly once. grid = (N/64, B), block = 256 (8 waves,
// each wave owns 32 output rows x 64 pixels).
// ---------------------------------------------------------------------------
__global__ __launch_bounds__(256) void k_qkv(
    const float* __restrict__ x,
    const float* __restrict__ Wq, const float* __restrict__ bq,
    const float* __restrict__ Wk, const float* __restrict__ bk,
    const float* __restrict__ Wv, const float* __restrict__ bv,
    __bf16* __restrict__ qT, __bf16* __restrict__ kT, __bf16* __restrict__ vT)
{
    __shared__ __attribute__((aligned(16))) __bf16 sA[256 * 48]; // W  [o][c]
    __shared__ __attribute__((aligned(16))) __bf16 sB[64 * 48];  // xT [n][c]

    const int n0 = blockIdx.x * 64;
    const int b  = blockIdx.y;
    const int t  = threadIdx.x;
    const int wave = t >> 5, lane = t & 31, half = lane >> 4, l16 = lane & 15;

    v8f acc[2][4];
#pragma unroll
    for (int i = 0; i < 2; ++i)
#pragma unroll
        for (int k = 0; k < 4; ++k)
#pragma unroll
            for (int j = 0; j < 8; ++j) acc[i][k][j] = 0.f;

#pragma unroll 1
    for (int c0 = 0; c0 < CDIM; c0 += 32) {
        { // stage weights: thread t = output row, 32 channels, f32->bf16 packed
            const float* src;
            if (t < 64)       src = Wq + (size_t)t * CDIM;
            else if (t < 128) src = Wk + (size_t)(t - 64) * CDIM;
            else              src = Wv + (size_t)(t - 128) * CDIM;
            const float4* s4 = (const float4*)(src + c0);
            float4 f[8];
#pragma unroll
            for (int i = 0; i < 8; ++i) f[i] = s4[i];
            unsigned u[16];
#pragma unroll
            for (int i = 0; i < 8; ++i) {
                u[2 * i]     = pk2f(f[i].x, f[i].y);
                u[2 * i + 1] = pk2f(f[i].z, f[i].w);
            }
            uint4* d = (uint4*)(sA + t * 48);
#pragma unroll
            for (int i = 0; i < 4; ++i) {
                uint4 q; q.x = u[4*i]; q.y = u[4*i+1]; q.z = u[4*i+2]; q.w = u[4*i+3];
                d[i] = q;
            }
        }
        { // stage x transposed, pair-packed: sB[n][kc..kc+1] <- x[kc..kc+1][n]
            int kc = (t >> 4) * 2;        // even channel pair 0..30
            int nb = (t & 15) * 4;        // 4 pixels
            const float* s0 = x + ((size_t)b * CDIM + c0 + kc) * NDIM + n0 + nb;
            float4 a = *(const float4*)s0;
            float4 c = *(const float4*)(s0 + NDIM);
            *(unsigned*)(sB + (nb + 0) * 48 + kc) = pk2f(a.x, c.x);
            *(unsigned*)(sB + (nb + 1) * 48 + kc) = pk2f(a.y, c.y);
            *(unsigned*)(sB + (nb + 2) * 48 + kc) = pk2f(a.z, c.z);
            *(unsigned*)(sB + (nb + 3) * 48 + kc) = pk2f(a.w, c.w);
        }
        __syncthreads();

        v16bf bb[4];
#pragma unroll
        for (int tt = 0; tt < 4; ++tt)
            bb[tt] = frag_b(sB + (tt * 16 + l16) * 48, half, 0);
#pragma unroll
        for (int rb = 0; rb < 2; ++rb) {
            v16bf a = frag_a(sA + (wave * 32 + rb * 16 + l16) * 48, half, 0);
#pragma unroll
            for (int tt = 0; tt < 4; ++tt)
                acc[rb][tt] = wmma_bf16(a, bb[tt], acc[rb][tt]);
        }
        __syncthreads();
    }

#pragma unroll
    for (int rb = 0; rb < 2; ++rb) {
        int ob = wave * 32 + rb * 16;            // wave-uniform strip base
        const float* bsrc; __bf16* dbase; int olocal;
        if (ob < 64)       { bsrc = bq; dbase = qT + (size_t)b * CQK * NDIM; olocal = ob; }
        else if (ob < 128) { bsrc = bk; dbase = kT + (size_t)b * CQK * NDIM; olocal = ob - 64; }
        else               { bsrc = bv; dbase = vT + (size_t)b * CV  * NDIM; olocal = ob - 128; }
#pragma unroll
        for (int r = 0; r < 8; ++r) {
            int o = olocal + r + 8 * half;
            float bias = bsrc[o];
            __bf16* dst = dbase + (size_t)o * NDIM + n0;
#pragma unroll
            for (int tt = 0; tt < 4; ++tt)
                dst[tt * 16 + l16] = (__bf16)(acc[rb][tt][r] + bias);
        }
    }
}

// ---------------------------------------------------------------------------
// Kernel 2: flash attention. Block = 64 query rows (4 waves x 16), streams
// 64-key tiles. S^T orientation keeps softmax stats per-lane (one shfl_xor).
// Q fragments hoisted out of the key loop; next tile prefetched.
// grid = (N/64, B), block = 128.
// ---------------------------------------------------------------------------
__global__ __launch_bounds__(128) void k_attn(
    const __bf16* __restrict__ qT, const __bf16* __restrict__ kT,
    const __bf16* __restrict__ vT, __bf16* __restrict__ ov)
{
    __shared__ __attribute__((aligned(16))) __bf16 sQ[64 * 80];    // [n][o]
    __shared__ __attribute__((aligned(16))) __bf16 sK[64 * 80];    // [m][o]
    __shared__ __attribute__((aligned(16))) __bf16 sV[128 * 80];   // [cv][m]
    __shared__ __attribute__((aligned(16))) __bf16 sP[4][16 * 80]; // per-wave [n][m]

    const int n0 = blockIdx.x * 64;
    const int b  = blockIdx.y;
    const int t  = threadIdx.x;
    const int wave = t >> 5, lane = t & 31, half = lane >> 4, l16 = lane & 15;

    // transpose-staging helpers: thread -> channel pair (2*op) x 16 columns
    const int op = t >> 2;            // 0..31 -> channels 2op, 2op+1
    const int mb = (t & 3) * 16;      // 16 columns

    { // stage Q (once): sQ[n][o..o+1] <- qT[o..o+1][n0+n], packed b32 writes
        const __bf16* s0 = qT + ((size_t)b * CQK + 2 * op) * NDIM + n0 + mb;
        const __bf16* s1 = s0 + NDIM;
        v8bf a0 = ((const v8bf*)s0)[0], a1 = ((const v8bf*)s0)[1];
        v8bf c0 = ((const v8bf*)s1)[0], c1 = ((const v8bf*)s1)[1];
#pragma unroll
        for (int i = 0; i < 8; ++i)
            *(unsigned*)(sQ + (mb + i) * 80 + 2 * op)     = pk2b(a0[i], c0[i]);
#pragma unroll
        for (int i = 0; i < 8; ++i)
            *(unsigned*)(sQ + (mb + 8 + i) * 80 + 2 * op) = pk2b(a1[i], c1[i]);
    }
    __syncthreads();

    // Q fragments are loop-invariant: hoist both K-chunks
    const __bf16* qrow = sQ + (wave * 16 + l16) * 80;
    v16bf bqf0 = frag_b(qrow, half, 0);
    v16bf bqf1 = frag_b(qrow, half, 32);

    v8f oacc[8];
#pragma unroll
    for (int i = 0; i < 8; ++i)
#pragma unroll
        for (int j = 0; j < 8; ++j) oacc[i][j] = 0.f;
    float mrow = -3.0e38f, lrow = 0.f;  // stats of query row n0+16*wave+l16

#pragma unroll 1
    for (int j = 0; j < NDIM; j += 64) {
        if (j + 64 < NDIM) { // prefetch next tile (global_prefetch_b8)
            __builtin_prefetch(kT + ((size_t)b * CQK + 2 * op) * NDIM + j + 64 + mb, 0, 0);
            __builtin_prefetch(vT + ((size_t)b * CV + t) * NDIM + j + 64, 0, 0);
        }
        { // stage K transposed, pair-packed: sK[m][o..o+1] <- kT[o..o+1][j+m]
            const __bf16* s0 = kT + ((size_t)b * CQK + 2 * op) * NDIM + j + mb;
            const __bf16* s1 = s0 + NDIM;
            v8bf a0 = ((const v8bf*)s0)[0], a1 = ((const v8bf*)s0)[1];
            v8bf c0 = ((const v8bf*)s1)[0], c1 = ((const v8bf*)s1)[1];
#pragma unroll
            for (int i = 0; i < 8; ++i)
                *(unsigned*)(sK + (mb + i) * 80 + 2 * op)     = pk2b(a0[i], c0[i]);
#pragma unroll
            for (int i = 0; i < 8; ++i)
                *(unsigned*)(sK + (mb + 8 + i) * 80 + 2 * op) = pk2b(a1[i], c1[i]);
        }
        { // stage V direct: sV[cv][m] <- vT[cv][j+m] (contiguous 128B rows)
            const uint4* src = (const uint4*)(vT + ((size_t)b * CV + t) * NDIM + j);
            uint4* dst = (uint4*)(sV + t * 80);
#pragma unroll
            for (int i = 0; i < 8; ++i) dst[i] = src[i];
        }
        __syncthreads();

        // S^T tiles: D[m][n] = sum_o K[m][o] * Q[n][o]
        v8f st[4];
#pragma unroll
        for (int i = 0; i < 4; ++i)
#pragma unroll
            for (int jj = 0; jj < 8; ++jj) st[i][jj] = 0.f;

#pragma unroll
        for (int ko = 0; ko < 64; ko += 32) {
            v16bf bqf = (ko == 0) ? bqf0 : bqf1;
            v16bf ak[4];
#pragma unroll
            for (int tt = 0; tt < 4; ++tt)
                ak[tt] = frag_a(sK + (tt * 16 + l16) * 80, half, ko);
#pragma unroll
            for (int tt = 0; tt < 4; ++tt)
                st[tt] = wmma_bf16(ak[tt], bqf, st[tt]);
        }

        // online softmax (per lane column n; halves hold different m rows)
        float vmax = -3.0e38f;
#pragma unroll
        for (int tt = 0; tt < 4; ++tt)
#pragma unroll
            for (int r = 0; r < 8; ++r) vmax = fmaxf(vmax, st[tt][r]);
        vmax = fmaxf(vmax, __shfl_xor(vmax, 16, 32));
        float mnew = fmaxf(mrow, vmax);
        float escl = __expf(mrow - mnew);
        float psum = 0.f;
#pragma unroll
        for (int tt = 0; tt < 4; ++tt)
#pragma unroll
            for (int r = 0; r < 8; ++r) {
                float p = __expf(st[tt][r] - mnew);
                st[tt][r] = p;
                psum += p;
            }
        psum += __shfl_xor(psum, 16, 32);
        lrow = lrow * escl + psum;
        mrow = mnew;

        // spill P^T -> sP[wave][n][m] (element r -> m = tt*16 + 8*half + r;
        // per-element b16 stores, wave-private buffer)
#pragma unroll
        for (int tt = 0; tt < 4; ++tt)
#pragma unroll
            for (int r = 0; r < 8; ++r)
                sP[wave][l16 * 80 + tt * 16 + r + 8 * half] = (__bf16)st[tt][r];

        // rescale O rows (row n = r + 8*half of this wave's block)
#pragma unroll
        for (int r = 0; r < 8; ++r) {
            float sc = __shfl(escl, r + 8 * half, 32);
#pragma unroll
            for (int c = 0; c < 8; ++c) oacc[c][r] *= sc;
        }

        // O[n][cv] += P[n][m] * V[cv][m]; batch B-fragments in groups of 4
        const __bf16* prow = sP[wave] + l16 * 80;
#pragma unroll
        for (int ko = 0; ko < 64; ko += 32) {
            v16bf ap = frag_a(prow, half, ko);
#pragma unroll
            for (int g = 0; g < 2; ++g) {
                v16bf bvf[4];
#pragma unroll
                for (int c = 0; c < 4; ++c)
                    bvf[c] = frag_b(sV + ((g * 4 + c) * 16 + l16) * 80, half, ko);
#pragma unroll
                for (int c = 0; c < 4; ++c)
                    oacc[g * 4 + c] = wmma_bf16(ap, bvf[c], oacc[g * 4 + c]);
            }
        }
        __syncthreads();
    }

    // finalize: divide by row sums, write out^T [n][cv] bf16
#pragma unroll
    for (int r = 0; r < 8; ++r) {
        float inv = 1.0f / __shfl(lrow, r + 8 * half, 32);
        int n = n0 + wave * 16 + r + 8 * half;
        __bf16* dst = ov + ((size_t)b * NDIM + n) * CV;
#pragma unroll
        for (int c = 0; c < 8; ++c)
            dst[c * 16 + l16] = (__bf16)(oacc[c][r] * inv);
    }
}

// ---------------------------------------------------------------------------
// Kernel 3: y = x + gamma * (Wo @ outT + bo).
// grid = (N/64, C/64, B), block = 128.
// ---------------------------------------------------------------------------
__global__ __launch_bounds__(128) void k_out(
    const float* __restrict__ x, const float* __restrict__ Wo,
    const float* __restrict__ bo, const float* __restrict__ gamma,
    const __bf16* __restrict__ ov, float* __restrict__ y)
{
    __shared__ __attribute__((aligned(16))) __bf16 sA[64 * 48]; // Wo [c][cv]
    __shared__ __attribute__((aligned(16))) __bf16 sB[64 * 48]; // ov [n][cv]

    const int n0 = blockIdx.x * 64;
    const int cs = blockIdx.y * 64;
    const int b  = blockIdx.z;
    const int t  = threadIdx.x;
    const int wave = t >> 5, lane = t & 31, half = lane >> 4, l16 = lane & 15;

    v8f acc[4];
#pragma unroll
    for (int i = 0; i < 4; ++i)
#pragma unroll
        for (int j = 0; j < 8; ++j) acc[i][j] = 0.f;

#pragma unroll 1
    for (int k0 = 0; k0 < CV; k0 += 32) {
        { // Wo chunk: 16 f32 per thread -> packed bf16, 2x uint4 LDS stores
            int row = t >> 1, seg = (t & 1) * 16;
            const float4* s4 = (const float4*)(Wo + (size_t)(cs + row) * CV + k0 + seg);
            float4 f[4];
#pragma unroll
            for (int i = 0; i < 4; ++i) f[i] = s4[i];
            unsigned u[8];
#pragma unroll
            for (int i = 0; i < 4; ++i) {
                u[2*i]   = pk2f(f[i].x, f[i].y);
                u[2*i+1] = pk2f(f[i].z, f[i].w);
            }
            uint4* d = (uint4*)(sA + row * 48 + seg);
            uint4 q0; q0.x = u[0]; q0.y = u[1]; q0.z = u[2]; q0.w = u[3];
            uint4 q1; q1.x = u[4]; q1.y = u[5]; q1.z = u[6]; q1.w = u[7];
            d[0] = q0; d[1] = q1;
        }
        { // ov chunk: contiguous bf16 copy, 2x uint4
            int row = t >> 1, seg = (t & 1) * 16;
            const uint4* src = (const uint4*)(ov + ((size_t)b * NDIM + n0 + row) * CV + k0 + seg);
            uint4* dst = (uint4*)(sB + row * 48 + seg);
            dst[0] = src[0]; dst[1] = src[1];
        }
        __syncthreads();

        v16bf bb[4];
#pragma unroll
        for (int tt = 0; tt < 4; ++tt)
            bb[tt] = frag_b(sB + (tt * 16 + l16) * 48, half, 0);
        v16bf a = frag_a(sA + (wave * 16 + l16) * 48, half, 0);
#pragma unroll
        for (int tt = 0; tt < 4; ++tt)
            acc[tt] = wmma_bf16(a, bb[tt], acc[tt]);
        __syncthreads();
    }

    float g = gamma[0];
#pragma unroll
    for (int r = 0; r < 8; ++r) {
        int c = cs + wave * 16 + r + 8 * half;
        float bias = bo[c];
        const size_t base = ((size_t)b * CDIM + c) * NDIM + n0;
#pragma unroll
        for (int tt = 0; tt < 4; ++tt) {
            size_t idx = base + tt * 16 + l16;
            y[idx] = x[idx] + g * (acc[tt][r] + bias);
        }
    }
}

// ---------------------------------------------------------------------------
extern "C" void kernel_launch(void* const* d_in, const int* in_sizes, int n_in,
                              void* d_out, int out_size, void* d_ws, size_t ws_size,
                              hipStream_t stream) {
    const float* x     = (const float*)d_in[0];
    const float* Wq    = (const float*)d_in[1];
    const float* bq    = (const float*)d_in[2];
    const float* Wk    = (const float*)d_in[3];
    const float* bk    = (const float*)d_in[4];
    const float* Wv    = (const float*)d_in[5];
    const float* bv    = (const float*)d_in[6];
    const float* Wo    = (const float*)d_in[7];
    const float* bo    = (const float*)d_in[8];
    const float* gamma = (const float*)d_in[9];

    const int B = in_sizes[0] / (CDIM * NDIM);   // = 4

    __bf16* qT = (__bf16*)d_ws;
    __bf16* kT = qT + (size_t)B * CQK * NDIM;
    __bf16* vT = kT + (size_t)B * CQK * NDIM;
    __bf16* ov = vT + (size_t)B * CV  * NDIM;     // total 12 MB of bf16 scratch

    k_qkv <<<dim3(NDIM / 64, B),            dim3(256), 0, stream>>>(x, Wq, bq, Wk, bk, Wv, bv, qT, kT, vT);
    k_attn<<<dim3(NDIM / 64, B),            dim3(128), 0, stream>>>(qT, kT, vT, ov);
    k_out <<<dim3(NDIM / 64, CDIM / 64, B), dim3(128), 0, stream>>>(x, Wo, bo, gamma, ov, (float*)d_out);
}
